// Contrast_4827543240964
// MI455X (gfx1250) — matrix operand order repaired and checked by reading, not compiled, not run on hardware
//
#include <hip/hip_runtime.h>
#include <hip/hip_bf16.h>

typedef _Float16 half_t;
typedef __attribute__((ext_vector_type(16))) _Float16 v16h;
typedef __attribute__((ext_vector_type(8)))  _Float16 v8h;
typedef __attribute__((ext_vector_type(8)))  float    v8f;

static constexpr int N = 16384;
static constexpr int D = 256;
static constexpr float INV_TAU = 5.0f;   // 1 / 0.2

// ---------------------------------------------------------------------------
// Kernel 1: row L2-normalize fp32 -> f16.  One wave32 per row (8 floats/lane).
// ---------------------------------------------------------------------------
__global__ __launch_bounds__(256) void nrm_kernel(const float* __restrict__ z,
                                                  half_t* __restrict__ h) {
    const int gtid = blockIdx.x * blockDim.x + threadIdx.x;
    const int row  = gtid >> 5;          // wave id == row
    const int lane = threadIdx.x & 31;

    const float* p = z + (size_t)row * D + lane * 8;
    float4 x0 = *(const float4*)(p);
    float4 x1 = *(const float4*)(p + 4);

    float ss = x0.x*x0.x + x0.y*x0.y + x0.z*x0.z + x0.w*x0.w
             + x1.x*x1.x + x1.y*x1.y + x1.z*x1.z + x1.w*x1.w;
    #pragma unroll
    for (int m = 16; m >= 1; m >>= 1) ss += __shfl_xor(ss, m, 32);

    const float inv = 1.0f / fmaxf(sqrtf(ss), 1e-12f);

    float v[8] = {x0.x, x0.y, x0.z, x0.w, x1.x, x1.y, x1.z, x1.w};
    v8h o;
    #pragma unroll
    for (int e = 0; e < 8; ++e) o[e] = (_Float16)(v[e] * inv);
    *(v8h*)(h + (size_t)row * D + lane * 8) = o;
}

// ---------------------------------------------------------------------------
// Kernel 2: WMMA GEMM h1 * h2^T over 128x128 blocks; per-element exp(sim/tau)
// folded immediately into per-row sums (never materializes the NxN matrix).
// 8 waves/block, each wave: 16 rows x 128 cols = 8 accumulators v8f.
// Double-buffered LDS staging (BMxBK + BNxBK f16 = 16KB per buffer).
// ---------------------------------------------------------------------------
__global__ __launch_bounds__(256) void gemm_rowsum_kernel(
        const half_t* __restrict__ h1, const half_t* __restrict__ h2,
        float* __restrict__ rowsum) {
    constexpr int BM = 128, BN = 128, BK = 32, KSTEPS = D / BK;

    __shared__ half_t lA[2][BM * BK];   // 8 KB per buffer
    __shared__ half_t lB[2][BN * BK];   // 8 KB per buffer

    const int bi   = blockIdx.y;
    const int bj   = blockIdx.x;
    const int tid  = threadIdx.x;
    const int wave = tid >> 5;
    const int lane = tid & 31;

    // ---- global->regs->LDS staging mapping: 2 threads per tile row,
    //      each thread moves 16 contiguous halfs (32 B) per matrix per step.
    const int ldRow = tid >> 1;
    const int ldSeg = (tid & 1) * 16;
    const half_t* gA = h1 + ((size_t)(bi * BM + ldRow)) * D + ldSeg;
    const half_t* gB = h2 + ((size_t)(bj * BN + ldRow)) * D + ldSeg;
    half_t* sA0 = &lA[0][ldRow * BK + ldSeg];
    half_t* sB0 = &lB[0][ldRow * BK + ldSeg];
    half_t* sA1 = &lA[1][ldRow * BK + ldSeg];
    half_t* sB1 = &lB[1][ldRow * BK + ldSeg];

    v8h ra0, ra1, rb0, rb1;

    // preload k-step 0 into buffer 0
    ra0 = *(const v8h*)(gA);     ra1 = *(const v8h*)(gA + 8);
    rb0 = *(const v8h*)(gB);     rb1 = *(const v8h*)(gB + 8);
    *(v8h*)(sA0) = ra0; *(v8h*)(sA0 + 8) = ra1;
    *(v8h*)(sB0) = rb0; *(v8h*)(sB0 + 8) = rb1;
    __syncthreads();

    // ---- WMMA fragment coordinates (ISA 7.12.2 wave32 layouts)
    const int mrow = lane & 15;          // A row / B col within 16-tile
    const int hf   = lane >> 4;          // lane half

    v8f acc[8] = {};                     // 8 N-tiles of 16x16 f32

    union AV { v16h v; v8h h[2]; };

    for (int ks = 0; ks < KSTEPS; ++ks) {
        const int cur = ks & 1;

        if (ks + 1 < KSTEPS) {           // fetch next k-slice into registers
            const int ko = (ks + 1) * BK;
            ra0 = *(const v8h*)(gA + ko);     ra1 = *(const v8h*)(gA + ko + 8);
            rb0 = *(const v8h*)(gB + ko);     rb1 = *(const v8h*)(gB + ko + 8);
        }

        // A fragment: lane (m, hf) holds 16 contiguous K at hf*16
        const half_t* aptr = &lA[cur][(wave * 16 + mrow) * BK];
        AV a;
        a.h[0] = *(const v8h*)(aptr + hf * 16);
        a.h[1] = *(const v8h*)(aptr + hf * 16 + 8);

        #pragma unroll
        for (int j = 0; j < 8; ++j) {
            // B fragment: lane (n, hf) holds 16 contiguous K at hf*16
            const half_t* bptr = &lB[cur][(j * 16 + mrow) * BK + hf * 16];
            v16h b = *(const v16h*)bptr;
            acc[j] = __builtin_amdgcn_wmma_f32_16x16x32_f16(
                         false, a.v, false, b, (short)0, acc[j], false, false);
        }

        if (ks + 1 < KSTEPS) {           // stage next slice into other buffer
            half_t* dA = cur ? sA0 : sA1;
            half_t* dB = cur ? sB0 : sB1;
            *(v8h*)(dA) = ra0; *(v8h*)(dA + 8) = ra1;
            *(v8h*)(dB) = rb0; *(v8h*)(dB + 8) = rb1;
            __syncthreads();
        }
    }

    // ---- per-row sum of exp(sim/tau) across this wave's 128 columns.
    // Acc layout: lane holds col n=mrow (per half), VGPR r -> row hf*8 + r.
    float part[8];
    #pragma unroll
    for (int r = 0; r < 8; ++r) {
        float s = 0.0f;
        #pragma unroll
        for (int j = 0; j < 8; ++j) s += __expf(acc[j][r] * INV_TAU);
        part[r] = s;
    }
    // reduce over the 16 lanes sharing one half (xor over lane bits 0..3)
    #pragma unroll
    for (int r = 0; r < 8; ++r) {
        #pragma unroll
        for (int m = 1; m < 16; m <<= 1) part[r] += __shfl_xor(part[r], m, 32);
    }
    if (mrow == 0) {                      // lanes 0 and 16
        const int rowBase = bi * BM + wave * 16 + hf * 8;
        #pragma unroll
        for (int r = 0; r < 8; ++r) atomicAdd(&rowsum[rowBase + r], part[r]);
    }
}

// ---------------------------------------------------------------------------
// Kernel 3: loss_i = log(rowsum_i - exp(s_i)) - s_i, s_i = dot(h1_i,h2_i)/tau.
// Block reduce + single atomicAdd into the scalar output.
// ---------------------------------------------------------------------------
__global__ __launch_bounds__(256) void finalize_kernel(
        const half_t* __restrict__ h1, const half_t* __restrict__ h2,
        const float* __restrict__ rowsum, float* __restrict__ out) {
    const int i = blockIdx.x * 256 + threadIdx.x;
    const half_t* p1 = h1 + (size_t)i * D;
    const half_t* p2 = h2 + (size_t)i * D;
    float dot = 0.0f;
    #pragma unroll 4
    for (int d = 0; d < D; d += 8) {
        v8h a = *(const v8h*)(p1 + d);
        v8h b = *(const v8h*)(p2 + d);
        #pragma unroll
        for (int e = 0; e < 8; ++e) dot += (float)a[e] * (float)b[e];
    }
    const float s  = dot * INV_TAU;
    const float li = logf(rowsum[i] - __expf(s)) - s;

    __shared__ float red[256];
    red[threadIdx.x] = li;
    __syncthreads();
    for (int st = 128; st > 0; st >>= 1) {
        if (threadIdx.x < st) red[threadIdx.x] += red[threadIdx.x + st];
        __syncthreads();
    }
    if (threadIdx.x == 0) atomicAdd(out, red[0]);
}

// ---------------------------------------------------------------------------
extern "C" void kernel_launch(void* const* d_in, const int* in_sizes, int n_in,
                              void* d_out, int out_size, void* d_ws, size_t ws_size,
                              hipStream_t stream) {
    const float* z1 = (const float*)d_in[0];
    const float* z2 = (const float*)d_in[1];
    float* out = (float*)d_out;

    half_t* h1     = (half_t*)d_ws;                 // N*D f16 = 8 MB
    half_t* h2     = h1 + (size_t)N * D;            // 8 MB
    float*  rowsum = (float*)(h2 + (size_t)N * D);  // 64 KB

    hipMemsetAsync(rowsum, 0, N * sizeof(float), stream);
    hipMemsetAsync(out, 0, sizeof(float), stream);

    // normalize: one wave per row -> N*32 threads
    nrm_kernel<<<(N * 32) / 256, 256, 0, stream>>>(z1, h1);
    nrm_kernel<<<(N * 32) / 256, 256, 0, stream>>>(z2, h2);

    dim3 grid(N / 128, N / 128);
    gemm_rowsum_kernel<<<grid, 256, 0, stream>>>(h1, h2, rowsum);

    finalize_kernel<<<N / 256, 256, 0, stream>>>(h1, h2, rowsum, out);
}
